// SlidingAttentionBlock_90443421319538
// MI455X (gfx1250) — matrix operand order repaired and checked by reading, not compiled, not run on hardware
//
#include <hip/hip_runtime.h>

// ---------------- problem constants (match reference) ----------------
#define B_      2
#define S_      2048
#define D_      2048
#define NH_     8
#define NKV_    4
#define HD_     256
#define FFN_    8192
#define CACHE_  4096
#define WINDOW_ 1024
#define EPS_    1e-6f
#define M_      (B_ * S_)          // 4096 activation rows

typedef __attribute__((ext_vector_type(16))) __bf16 v16bf;
typedef __attribute__((ext_vector_type(8)))  float  v8f;

static __device__ __forceinline__ __bf16 f2bf(float f) { return (__bf16)f; }

// ---------------- CDNA5 data-movement paths ----------------
#if defined(__has_builtin)
# if __has_builtin(__builtin_amdgcn_tensor_load_to_lds)
#  define HAVE_TDM 1
# endif
#endif
#ifndef HAVE_TDM
# define HAVE_TDM 0
#endif

// async global->LDS (GLOBAL_LOAD_ASYNC_TO_LDS_B128, ASYNCcnt)
static __device__ __forceinline__ void stage_b128(const void* g, void* l) {
    unsigned la = (unsigned)(unsigned long long)l;
    asm volatile("global_load_async_to_lds_b128 %0, %1, off"
                 :: "v"(la), "v"(g) : "memory");
}

#if HAVE_TDM
typedef __attribute__((ext_vector_type(4))) unsigned tdm_g0_t;
typedef __attribute__((ext_vector_type(8))) int      tdm_g1_t;
typedef __attribute__((ext_vector_type(4))) int      tdm_g2_t;

// 2D TDM load: tile (rows x 32 bf16), global row stride = strideK elements,
// LDS padded 2 DWORDs after every 16 DWORDs (-> 36-element LDS rows).
static __device__ __forceinline__ void tdm_load_tile(const __bf16* g, void* lds,
                                                     int rows, int strideK)
{
    const unsigned long long ga = (unsigned long long)g;
    tdm_g0_t g0;
    g0.x = 1u;                                            // count=1 (valid D#)
    g0.y = (unsigned)(unsigned long long)lds;             // lds_addr
    g0.z = (unsigned)(ga & 0xFFFFFFFFu);                  // global_addr[31:0]
    g0.w = (unsigned)((ga >> 32) & 0x1FFFFFFu) | (2u << 30); // addr[56:32] | type=2
    tdm_g1_t g1;
    g1[0] = (int)((1u << 16)      // data_size = 2 bytes
                | (1u << 20)      // pad_enable
                | (3u << 22)      // pad_interval = 16 DWORDs
                | (1u << 25));    // pad_amount  = 2 DWORDs
    g1[1] = (int)(32u << 16);                 // tensor_dim0 = 32
    g1[2] = (int)((unsigned)rows << 16);      // tensor_dim1 = rows
    g1[3] = (int)(32u << 16);                 // tile_dim0 = 32
    g1[4] = rows;                             // tile_dim1 = rows
    g1[5] = strideK;                          // tensor_dim0_stride[31:0]
    g1[6] = 0;
    g1[7] = 0;
    tdm_g2_t z4 = {0, 0, 0, 0};
    tdm_g1_t z8 = {0, 0, 0, 0, 0, 0, 0, 0};
    __builtin_amdgcn_tensor_load_to_lds(g0, g1, z4, z4, z8, 0);
}
# define ASYNC_PENDING "0x4"       // 4 older B-tile asyncs outstanding allowed
#else
# define ASYNC_PENDING "0x6"       // A(2) + B(4) per stage
#endif

// =====================================================================
// WMMA GEMM, all-bf16 operands: C[M,N] = A[M,K] * Bt[N,K]^T
// A row-major bf16, Bt pre-transposed [N][K] bf16.
// Block 128x256, K-step 32, double-buffered LDS; A tile staged by the
// Tensor Data Mover (wave 0), B tile by async global->LDS; 8 waves,
// wave tile 64x64 = 16 WMMAs per k-step.  M%128==0, N%256==0, K%32==0.
// =====================================================================
template <typename OutT>
__launch_bounds__(256)
__global__ void wmma_gemm_kernel(const __bf16* __restrict__ A,
                                 const __bf16* __restrict__ Bt,
                                 OutT* __restrict__ C,
                                 int M, int N, int K)
{
    __shared__ __bf16 lsA[2][128][36];
    __shared__ __bf16 lsB[2][256][36];

    const int tid = threadIdx.x;
    const int bm = blockIdx.y * 128;
    const int bn = blockIdx.x * 256;
    const int lane = tid & 31, wid = tid >> 5;
    const int laneRow = lane & 15, hi = lane >> 4;
    const int wm = (wid >> 2) * 64;        // 0,64
    const int wn = (wid & 3) * 64;         // 0,64,128,192
    const int ar = tid >> 1, ac = (tid & 1) * 16;

    v8f acc[4][4];
#pragma unroll
    for (int i = 0; i < 4; ++i)
#pragma unroll
        for (int j = 0; j < 4; ++j)
            acc[i][j] = (v8f){0.f,0.f,0.f,0.f,0.f,0.f,0.f,0.f};

    auto stageB = [&](int buf, int k0) {
        const __bf16* s = Bt + (size_t)(bn + tid) * K + k0;
        __bf16* d = &lsB[buf][tid][0];
        stage_b128(s,      d);
        stage_b128(s + 8,  d + 8);
        stage_b128(s + 16, d + 16);
        stage_b128(s + 24, d + 24);
    };
    auto stageA = [&](int buf, int k0) {
#if HAVE_TDM
        if (wid == 0)
            tdm_load_tile(A + (size_t)bm * K + k0, &lsA[buf][0][0], 128, K);
#else
        const __bf16* s = A + (size_t)(bm + ar) * K + k0 + ac;
        __bf16* d = &lsA[buf][ar][ac];
        stage_b128(s,     d);
        stage_b128(s + 8, d + 8);
#endif
    };

    stageA(0, 0);
    stageB(0, 0);
    int buf = 0;

    for (int k0 = 0; k0 < K; k0 += 32) {
        const bool more = (k0 + 32) < K;
        if (more) {                         // overlap next tile's DMA with compute
            stageA(buf ^ 1, k0 + 32);
            stageB(buf ^ 1, k0 + 32);
            asm volatile("s_wait_asynccnt " ASYNC_PENDING ::: "memory");
#if HAVE_TDM
            if (wid == 0) __builtin_amdgcn_s_wait_tensorcnt(1);
#endif
        } else {
            asm volatile("s_wait_asynccnt 0x0" ::: "memory");
#if HAVE_TDM
            if (wid == 0) __builtin_amdgcn_s_wait_tensorcnt(0);
#endif
        }
        __syncthreads();

        v16bf af[4], bfv[4];
#pragma unroll
        for (int mi = 0; mi < 4; ++mi) {
            const __bf16* rp = &lsA[buf][wm + mi * 16 + laneRow][0];
#pragma unroll
            for (int v = 0; v < 8; ++v) {
                // A 16x32 bf16 layout: lanes0-15 K={0..7,16..23}, lanes16-31 K={8..15,24..31}
                const int kb = (v >> 2) * 16 + hi * 8 + (v & 3) * 2;
                af[mi][2 * v]     = rp[kb];
                af[mi][2 * v + 1] = rp[kb + 1];
            }
        }
#pragma unroll
        for (int ni = 0; ni < 4; ++ni) {
            const __bf16* cp = &lsB[buf][wn + ni * 16 + laneRow][0];
#pragma unroll
            for (int v = 0; v < 8; ++v) {
                // B 32x16: lanes0-15 K={0..7,16..23}, lanes16-31 K={8..15,24..31}
                const int kb = (v >> 2) * 16 + hi * 8 + (v & 3) * 2;
                bfv[ni][2 * v]     = cp[kb];
                bfv[ni][2 * v + 1] = cp[kb + 1];
            }
        }
#pragma unroll
        for (int mi = 0; mi < 4; ++mi)
#pragma unroll
            for (int ni = 0; ni < 4; ++ni)
                acc[mi][ni] = __builtin_amdgcn_wmma_f32_16x16x32_bf16(
                    false, af[mi], false, bfv[ni], (short)0, acc[mi][ni],
                    false, false);
        __syncthreads();
        buf ^= 1;
    }

#pragma unroll
    for (int mi = 0; mi < 4; ++mi)
#pragma unroll
        for (int ni = 0; ni < 4; ++ni) {
            const int col = bn + wn + ni * 16 + laneRow;
#pragma unroll
            for (int r = 0; r < 8; ++r) {
                const int row = bm + wm + mi * 16 + hi * 8 + r;
                C[(size_t)row * N + col] = (OutT)acc[mi][ni][r];
            }
        }
}

// =====================================================================
// Wave-local WMMA flash attention (sliding window, unscaled scores).
// One wave = 16 queries; block = 8 waves; Q,K bf16 row-major;
// V pre-transposed bf16 [HD][CACHE].
// =====================================================================
__launch_bounds__(256)
__global__ void flash_attn_kernel(const __bf16* __restrict__ qbf,  // [B,S,NH,HD]
                                  const __bf16* __restrict__ kbf,  // [B,NKV,CACHE,HD]
                                  const __bf16* __restrict__ vtbf, // [B,NKV,HD,CACHE]
                                  __bf16* __restrict__ attn)       // [B,S,NH*HD]
{
    __shared__ __bf16 lsP[8][16][36];          // per-wave P staging

    const int tid = threadIdx.x;
    const int wid = tid >> 5, lane = tid & 31;
    const int laneRow = lane & 15, hi = lane >> 4;

    const int blk = blockIdx.x;                // (b, h, qt)
    const int qt = blk % (S_ / 128);
    const int h  = (blk / (S_ / 128)) % NH_;
    const int b  = blk / ((S_ / 128) * NH_);
    const int q0 = qt * 128 + wid * 16;
    const int kv = h / (NH_ / NKV_);

    const __bf16* Qb = qbf + (((size_t)b * S_ + q0) * NH_ + h) * HD_;
    const __bf16* Kb = kbf + ((size_t)b * NKV_ + kv) * (size_t)CACHE_ * HD_;
    const __bf16* Vt = vtbf + ((size_t)b * NKV_ + kv) * (size_t)HD_ * CACHE_;

    // resident Q fragments (8 hd-chunks of 32)
    v16bf qf[8];
#pragma unroll
    for (int c = 0; c < 8; ++c) {
        const __bf16* qrow = Qb + (size_t)laneRow * (NH_ * HD_) + c * 32;
#pragma unroll
        for (int v = 0; v < 8; ++v) {
            const int kb = hi * 16 + 2 * v;
            qf[c][2 * v]     = qrow[kb];
            qf[c][2 * v + 1] = qrow[kb + 1];
        }
    }

    v8f out[16];
#pragma unroll
    for (int t = 0; t < 16; ++t) out[t] = (v8f){0.f,0.f,0.f,0.f,0.f,0.f,0.f,0.f};
    float mrow[8], lrow[8];
#pragma unroll
    for (int r = 0; r < 8; ++r) { mrow[r] = -1e30f; lrow[r] = 0.f; }

    const int qmax = q0 + 15;
    int kmin = q0 - (WINDOW_ - 1);
    if (kmin < 0) kmin = 0;
    const int kb0 = kmin & ~31;

    for (int kb = kb0; kb <= qmax; kb += 32) {
        // ---- scores: 2 key-tiles x 8 hd-chunks ----
        v8f sc[2];
#pragma unroll
        for (int kt = 0; kt < 2; ++kt) {
            sc[kt] = (v8f){0.f,0.f,0.f,0.f,0.f,0.f,0.f,0.f};
            const __bf16* krow = Kb + (size_t)(kb + kt * 16 + laneRow) * HD_;
#pragma unroll
            for (int c = 0; c < 8; ++c) {
                v16bf kf = *(const v16bf*)(krow + c * 32 + hi * 16);
                sc[kt] = __builtin_amdgcn_wmma_f32_16x16x32_bf16(
                    false, qf[c], false, kf, (short)0, sc[kt], false, false);
            }
        }

        // ---- mask + online softmax (row stats across each half-wave) ----
        const int k0e = kb + laneRow;
        const int k1e = kb + 16 + laneRow;
        float scale[8];
#pragma unroll
        for (int r = 0; r < 8; ++r) {
            const int q = q0 + r + 8 * hi;
            const bool ok0 = (k0e <= q) && (q - k0e < WINDOW_);
            const bool ok1 = (k1e <= q) && (q - k1e < WINDOW_);
            float s0 = ok0 ? sc[0][r] : -1e30f;
            float s1 = ok1 ? sc[1][r] : -1e30f;
            float rm = fmaxf(s0, s1);
#pragma unroll
            for (int off = 1; off < 16; off <<= 1) rm = fmaxf(rm, __shfl_xor(rm, off));
            const float mnew = fmaxf(mrow[r], rm);
            scale[r] = __expf(mrow[r] - mnew);
            mrow[r] = mnew;
            const float p0 = ok0 ? __expf(sc[0][r] - mnew) : 0.f;
            const float p1 = ok1 ? __expf(sc[1][r] - mnew) : 0.f;
            sc[0][r] = p0;
            sc[1][r] = p1;
            float rs = p0 + p1;
#pragma unroll
            for (int off = 1; off < 16; off <<= 1) rs += __shfl_xor(rs, off);
            lrow[r] = lrow[r] * scale[r] + rs;
        }
#pragma unroll
        for (int t = 0; t < 16; ++t)
#pragma unroll
            for (int r = 0; r < 8; ++r) out[t][r] *= scale[r];

        // ---- stage P (C-layout) into A-frag layout via per-wave LDS ----
        __bf16 (*P)[36] = lsP[wid];
#pragma unroll
        for (int kt = 0; kt < 2; ++kt)
#pragma unroll
            for (int r = 0; r < 8; ++r)
                P[r + 8 * hi][kt * 16 + laneRow] = f2bf(sc[kt][r]);
        v16bf pf;
        {
            const __bf16* pr = &P[laneRow][0];
#pragma unroll
            for (int v = 0; v < 8; ++v) {
                const int kb2 = hi * 16 + 2 * v;
                pf[2 * v]     = pr[kb2];
                pf[2 * v + 1] = pr[kb2 + 1];
            }
        }

        // ---- P x V: 16 hd-tiles, contiguous V fragments from Vt ----
#pragma unroll
        for (int ht = 0; ht < 16; ++ht) {
            v16bf vf = *(const v16bf*)(Vt + (size_t)(ht * 16 + laneRow) * CACHE_
                                       + kb + hi * 16);
            out[ht] = __builtin_amdgcn_wmma_f32_16x16x32_bf16(
                false, pf, false, vf, (short)0, out[ht], false, false);
        }
    }

    // ---- epilogue ----
#pragma unroll
    for (int ht = 0; ht < 16; ++ht)
#pragma unroll
        for (int r = 0; r < 8; ++r) {
            const size_t row = (size_t)b * S_ + q0 + r + 8 * hi;
            attn[(row * NH_ + h) * HD_ + ht * 16 + laneRow] =
                f2bf(out[ht][r] / lrow[r]);
        }
}

// =====================================================================
// RMSNorm rows of 2048 with fused residual-add / scalar epilogue.
// =====================================================================
template <typename OutT>
__launch_bounds__(256)
__global__ void rmsnorm2048_kernel(const float* __restrict__ in,
                                   const float* __restrict__ w,
                                   const float* __restrict__ residual,
                                   const float* __restrict__ scale,
                                   OutT* __restrict__ out)
{
    __shared__ float red[8];
    const size_t row = blockIdx.x;
    const int tid = threadIdx.x;
    const float* r = in + row * (size_t)D_;

    float v[8];
    float ss = 0.f;
#pragma unroll
    for (int i = 0; i < 8; ++i) { v[i] = r[tid + 256 * i]; ss += v[i] * v[i]; }
#pragma unroll
    for (int off = 16; off > 0; off >>= 1) ss += __shfl_down(ss, off);
    if ((tid & 31) == 0) red[tid >> 5] = ss;
    __syncthreads();
    float tot = 0.f;
#pragma unroll
    for (int i = 0; i < 8; ++i) tot += red[i];
    const float inv = rsqrtf(tot / (float)D_ + EPS_);
    const float sc = scale ? scale[0] : 1.f;

#pragma unroll
    for (int i = 0; i < 8; ++i) {
        const int c = tid + 256 * i;
        float y = v[i] * inv;
        if (w) y *= w[c];
        if (residual) y += residual[row * (size_t)D_ + c];
        out[row * (size_t)D_ + c] = (OutT)(y * sc);
    }
}

// =====================================================================
// Per-head RMSNorm (+weight) + optional RoPE; out to [B,S,H,HD] or the
// KV cache [B,H,CACHE,HD] at seqpos[b]+s.
// =====================================================================
template <typename OutT>
__launch_bounds__(128)
__global__ void head_norm_rope_kernel(const float* __restrict__ in,
                                      const float* __restrict__ w,
                                      const float* __restrict__ freqs,
                                      const int* __restrict__ seqpos,
                                      OutT* __restrict__ out,
                                      int H, int do_rope, int cache_mode)
{
    __shared__ float red[4];
    const int idx = blockIdx.x;
    const int h = idx % H;
    const int s = (idx / H) % S_;
    const int b = idx / (H * S_);
    const int tid = threadIdx.x;

    const float* rowp = in + (((size_t)b * S_ + s) * H + h) * HD_;
    float x0 = rowp[2 * tid];
    float x1 = rowp[2 * tid + 1];

    float ss = x0 * x0 + x1 * x1;
#pragma unroll
    for (int off = 16; off > 0; off >>= 1) ss += __shfl_down(ss, off);
    if ((tid & 31) == 0) red[tid >> 5] = ss;
    __syncthreads();
    const float inv = rsqrtf((red[0] + red[1] + red[2] + red[3]) / (float)HD_ + EPS_);
    x0 *= inv;
    x1 *= inv;
    if (w) { x0 *= w[2 * tid]; x1 *= w[2 * tid + 1]; }

    const int pos = seqpos[b] + s;
    if (do_rope) {
        const float* fc = freqs + ((size_t)pos * (HD_ / 2) + tid) * 2;
        const float c = fc[0], sn = fc[1];
        const float re = x0 * c - x1 * sn;
        const float im = x0 * sn + x1 * c;
        x0 = re; x1 = im;
    }

    OutT* o;
    if (cache_mode)
        o = out + (((size_t)b * H + h) * CACHE_ + pos) * HD_ + 2 * tid;
    else
        o = out + (((size_t)b * S_ + s) * H + h) * HD_ + 2 * tid;
    o[0] = (OutT)x0;
    o[1] = (OutT)x1;
}

// Copy untouched cache region (positions outside [pos, pos+S)).
__launch_bounds__(256)
__global__ void cache_copy_kernel(const float* __restrict__ src,
                                  float* __restrict__ dst,
                                  const int* __restrict__ seqpos)
{
    const size_t idx = (size_t)blockIdx.x * 256 + threadIdx.x;
    const size_t total = (size_t)B_ * NKV_ * CACHE_ * HD_;
    if (idx >= total) return;
    const int p = (int)((idx / HD_) % CACHE_);
    const int b = (int)(idx / ((size_t)NKV_ * CACHE_ * HD_));
    const int pos = seqpos[b];
    if (p >= pos && p < pos + S_) return;
    dst[idx] = src[idx];
}

// f32 -> bf16 elementwise (n multiple of 4).
__launch_bounds__(256)
__global__ void cvt_bf16_kernel(const float* __restrict__ in,
                                __bf16* __restrict__ out, size_t n4)
{
    const size_t i = (size_t)blockIdx.x * 256 + threadIdx.x;
    if (i >= n4) return;
    const float4 f = ((const float4*)in)[i];
    const size_t b = i * 4;
    out[b + 0] = f2bf(f.x);
    out[b + 1] = f2bf(f.y);
    out[b + 2] = f2bf(f.z);
    out[b + 3] = f2bf(f.w);
}

// Batched transpose+convert: in [batch][R][C] f32 -> out [batch][C][R] bf16.
__launch_bounds__(256)
__global__ void transpose_cvt_kernel(const float* __restrict__ in,
                                     __bf16* __restrict__ out,
                                     int R, int C)
{
    __shared__ float t[32][33];
    const int bc = blockIdx.x * 32;
    const int br = blockIdx.y * 32;
    const size_t batch = blockIdx.z;
    const float* src = in + batch * (size_t)R * C;
    __bf16* dst = out + batch * (size_t)R * C;
    const int tc = threadIdx.x & 31;
    const int tr = threadIdx.x >> 5;
#pragma unroll
    for (int i = 0; i < 32; i += 8)
        t[tr + i][tc] = src[(size_t)(br + tr + i) * C + bc + tc];
    __syncthreads();
    const int rr = threadIdx.x & 31;
    const int cc = threadIdx.x >> 5;
#pragma unroll
    for (int i = 0; i < 32; i += 8)
        dst[(size_t)(bc + cc + i) * R + br + rr] = f2bf(t[rr][cc + i]);
}

// gelu(gate)*up, bf16 in/out (tanh approximation, as jax.nn.gelu default).
__launch_bounds__(256)
__global__ void gelu_mul_kernel(__bf16* __restrict__ gate,
                                const __bf16* __restrict__ up, size_t n)
{
    const size_t i = (size_t)blockIdx.x * 256 + threadIdx.x;
    if (i >= n) return;
    const float g = (float)gate[i];
    const float u = (float)up[i];
    const float t = 0.7978845608028654f * (g + 0.044715f * g * g * g);
    gate[i] = f2bf(0.5f * g * (1.f + tanhf(t)) * u);
}

// =====================================================================
// Host-side orchestration
// =====================================================================
extern "C" void kernel_launch(void* const* d_in, const int* in_sizes, int n_in,
                              void* d_out, int out_size, void* d_ws, size_t ws_size,
                              hipStream_t stream)
{
    (void)in_sizes; (void)n_in; (void)out_size; (void)ws_size;

    const float* x        = (const float*)d_in[0];
    const float* freqs    = (const float*)d_in[1];
    const float* k_cache  = (const float*)d_in[2];
    const float* v_cache  = (const float*)d_in[3];
    const int*   seqpos   = (const int*)  d_in[4];
    const float* wq       = (const float*)d_in[6];
    const float* wk       = (const float*)d_in[7];
    const float* wv       = (const float*)d_in[8];
    const float* wo       = (const float*)d_in[9];
    const float* q_norm   = (const float*)d_in[10];
    const float* k_norm   = (const float*)d_in[11];
    const float* w_gate   = (const float*)d_in[12];
    const float* w_up     = (const float*)d_in[13];
    const float* w_down   = (const float*)d_in[14];
    const float* ln_in    = (const float*)d_in[15];
    const float* ln_post  = (const float*)d_in[16];
    const float* ln_pre_f = (const float*)d_in[17];
    const float* ln_post_f= (const float*)d_in[18];
    const float* lscalar  = (const float*)d_in[19];

    float* out_x = (float*)d_out;
    float* out_k = out_x + (size_t)M_ * D_;
    float* out_v = out_k + (size_t)B_ * NKV_ * CACHE_ * HD_;

    // ---------------- workspace carve-up ----------------
    float* o_q = (float*)d_ws;                       // [M,2048] q f32 / proj / ffnd
    float* o_k = o_q + (size_t)M_ * 2048;            // [M,1024]
    float* o_v = o_k + (size_t)M_ * 1024;            // [M,1024]
    float* o_x1   = o_k;                             // [M,2048] reuse k+v region
    float* o_proj = o_q;
    float* o_ffnd = o_q;

    __bf16* pb = (__bf16*)(o_v + (size_t)M_ * 1024);
    __bf16* hbf    = pb; pb += (size_t)M_ * 2048;            // h / hffn
    __bf16* qbf    = pb; pb += (size_t)M_ * 2048;
    __bf16* kbf    = pb; pb += (size_t)B_ * NKV_ * CACHE_ * HD_;
    __bf16* vtbf   = pb; pb += (size_t)B_ * NKV_ * CACHE_ * HD_;
    __bf16* attnbf = pb; pb += (size_t)M_ * 2048;
    __bf16* gatebf = pb; pb += (size_t)M_ * FFN_;
    __bf16* upbf   = pb; pb += (size_t)M_ * FFN_;
    __bf16* wqb = pb; pb += (size_t)D_ * 2048;       // all weights [N][K]
    __bf16* wkb = pb; pb += (size_t)D_ * 1024;
    __bf16* wvb = pb; pb += (size_t)D_ * 1024;
    __bf16* wob = pb; pb += (size_t)2048 * D_;
    __bf16* wgb = pb; pb += (size_t)D_ * FFN_;
    __bf16* wub = pb; pb += (size_t)D_ * FFN_;
    __bf16* wdb = pb; pb += (size_t)FFN_ * D_;

    const dim3 blk256(256), blk128(128);

    // 0) weight transpose+convert: W[K][N] f32 -> Wt[N][K] bf16
    transpose_cvt_kernel<<<dim3(2048/32, 2048/32, 1), blk256, 0, stream>>>(wq, wqb, 2048, 2048);
    transpose_cvt_kernel<<<dim3(1024/32, 2048/32, 1), blk256, 0, stream>>>(wk, wkb, 2048, 1024);
    transpose_cvt_kernel<<<dim3(1024/32, 2048/32, 1), blk256, 0, stream>>>(wv, wvb, 2048, 1024);
    transpose_cvt_kernel<<<dim3(2048/32, 2048/32, 1), blk256, 0, stream>>>(wo, wob, 2048, 2048);
    transpose_cvt_kernel<<<dim3(FFN_/32, 2048/32, 1), blk256, 0, stream>>>(w_gate, wgb, 2048, FFN_);
    transpose_cvt_kernel<<<dim3(FFN_/32, 2048/32, 1), blk256, 0, stream>>>(w_up,   wub, 2048, FFN_);
    transpose_cvt_kernel<<<dim3(2048/32, FFN_/32, 1), blk256, 0, stream>>>(w_down, wdb, FFN_, 2048);

    // 1) h = rmsnorm(x, input_layernorm) -> bf16
    rmsnorm2048_kernel<__bf16><<<M_, blk256, 0, stream>>>(x, ln_in, nullptr, nullptr, hbf);

    // 2-4) QKV projections (f32 out for norm/rope stage)
    wmma_gemm_kernel<float><<<dim3(2048/256, M_/128), blk256, 0, stream>>>(hbf, wqb, o_q, M_, 2048, D_);
    wmma_gemm_kernel<float><<<dim3(1024/256, M_/128), blk256, 0, stream>>>(hbf, wkb, o_k, M_, 1024, D_);
    wmma_gemm_kernel<float><<<dim3(1024/256, M_/128), blk256, 0, stream>>>(hbf, wvb, o_v, M_, 1024, D_);

    // 5) q: head RMSNorm + RoPE -> bf16 [B,S,NH,HD]
    head_norm_rope_kernel<__bf16><<<B_*S_*NH_, blk128, 0, stream>>>(
        o_q, q_norm, freqs, seqpos, qbf, NH_, 1, 0);
    // 6) k: head RMSNorm + RoPE -> f32 keys_c (d_out)
    head_norm_rope_kernel<float><<<B_*S_*NKV_, blk128, 0, stream>>>(
        o_k, k_norm, freqs, seqpos, out_k, NKV_, 1, 1);
    // 7) v: head RMSNorm (no scale/rope) -> f32 values_c (d_out)
    head_norm_rope_kernel<float><<<B_*S_*NKV_, blk128, 0, stream>>>(
        o_v, nullptr, freqs, seqpos, out_v, NKV_, 0, 1);
    // 8) passthrough of untouched cache region
    {
        const int total = B_ * NKV_ * CACHE_ * HD_;
        cache_copy_kernel<<<(total + 255)/256, blk256, 0, stream>>>(k_cache, out_k, seqpos);
        cache_copy_kernel<<<(total + 255)/256, blk256, 0, stream>>>(v_cache, out_v, seqpos);
    }
    // 8b) bf16 copies of the full caches for the WMMA attention
    {
        const size_t n = (size_t)B_ * NKV_ * CACHE_ * HD_;
        cvt_bf16_kernel<<<(unsigned)((n/4 + 255)/256), blk256, 0, stream>>>(out_k, kbf, n/4);
        transpose_cvt_kernel<<<dim3(HD_/32, CACHE_/32, B_*NKV_), blk256, 0, stream>>>(out_v, vtbf, CACHE_, HD_);
    }

    // 9) WMMA flash attention -> attnbf [M, NH*HD] bf16
    flash_attn_kernel<<<B_ * NH_ * (S_/128), blk256, 0, stream>>>(qbf, kbf, vtbf, attnbf);

    // 10) proj = attn @ wo
    wmma_gemm_kernel<float><<<dim3(2048/256, M_/128), blk256, 0, stream>>>(attnbf, wob, o_proj, M_, D_, 2048);

    // 11) x1 = x + rmsnorm(proj, post_attention_layernorm)
    rmsnorm2048_kernel<float><<<M_, blk256, 0, stream>>>(o_proj, ln_post, x, nullptr, o_x1);

    // 12) hffn = rmsnorm(x1, pre_feedforward_layernorm) -> bf16
    rmsnorm2048_kernel<__bf16><<<M_, blk256, 0, stream>>>(o_x1, ln_pre_f, nullptr, nullptr, hbf);

    // 13-14) gate / up (bf16 out)
    wmma_gemm_kernel<__bf16><<<dim3(FFN_/256, M_/128), blk256, 0, stream>>>(hbf, wgb, gatebf, M_, FFN_, D_);
    wmma_gemm_kernel<__bf16><<<dim3(FFN_/256, M_/128), blk256, 0, stream>>>(hbf, wub, upbf,   M_, FFN_, D_);

    // 15) gate = gelu(gate) * up
    {
        const size_t n = (size_t)M_ * FFN_;
        gelu_mul_kernel<<<(unsigned)((n + 255)/256), blk256, 0, stream>>>(gatebf, upbf, n);
    }

    // 16) ffnd = act @ w_down
    wmma_gemm_kernel<float><<<dim3(2048/256, M_/128), blk256, 0, stream>>>(gatebf, wdb, o_ffnd, M_, D_, FFN_);

    // 17) out = (x1 + rmsnorm(ffnd, post_ffn_layernorm)) * layer_scalar
    rmsnorm2048_kernel<float><<<M_, blk256, 0, stream>>>(o_ffnd, ln_post_f, o_x1, lscalar, out_x);
}
